// RingAttention_50852412785065
// MI455X (gfx1250) — matrix operand order repaired and checked by reading, compiled, not run on hardware
//
#include <hip/hip_runtime.h>
#include <hip/hip_fp16.h>
#include <math.h>

typedef __attribute__((ext_vector_type(16))) _Float16 v16h;
typedef __attribute__((ext_vector_type(8)))  float    v8f;

#define S_LEN 2048
#define NHEAD 32
#define HDIM  128
#define QTILE 64
#define KTILE 32
#define LDK   (HDIM + 4)    // K tile row stride (halves): rows 264B -> bank step 2/row
#define LDV   (KTILE + 4)   // transposed-V row stride (halves): rows 72B (8B-aligned), bank step 18/row
#define LDP   (KTILE + 4)   // P scratch row stride (halves)

union F16Frag { v16h v; _Float16 e[16]; };
union F32Frag { v8f  v; float    e[8];  };

// Online-softmax update for one 16x32 score tile (two 16x16 C fragments).
// MASK instantiation only used on diagonal tiles; off-diagonal tiles skip all compares.
template<bool MASK>
__device__ __forceinline__ void online_update(
    F32Frag& s0, F32Frag& s1, float* mrow, float* lrow, F32Frag* o,
    _Float16* pw, int qr0, int jb, int hi, int ln)
{
    #pragma unroll
    for (int r = 0; r < 8; ++r) {
        float v0 = s0.e[r], v1 = s1.e[r];
        if (MASK) {
            int qg = qr0 + hi * 8 + r;
            if (jb + ln      > qg) v0 = -INFINITY;
            if (jb + 16 + ln > qg) v1 = -INFINITY;
        }
        float tmax = fmaxf(v0, v1);
        #pragma unroll
        for (int d = 1; d < 16; d <<= 1)
            tmax = fmaxf(tmax, __shfl_xor(tmax, d, 32));
        float mnew  = fmaxf(mrow[r], tmax);
        float alpha = exp2f(mrow[r] - mnew);   // == 0 on first tile (mrow = -inf)
        float p0 = exp2f(v0 - mnew);
        float p1 = exp2f(v1 - mnew);
        float rs = p0 + p1;
        #pragma unroll
        for (int d = 1; d < 16; d <<= 1)
            rs += __shfl_xor(rs, d, 32);
        lrow[r] = lrow[r] * alpha + rs;
        mrow[r] = mnew;
        #pragma unroll
        for (int t = 0; t < 8; ++t) o[t].e[r] *= alpha;
        pw[(hi * 8 + r) * LDP + ln]      = (_Float16)p0;
        pw[(hi * 8 + r) * LDP + 16 + ln] = (_Float16)p1;
    }
}

__global__ __launch_bounds__(128) void fa_fwd_kernel(
    const float* __restrict__ Q, const float* __restrict__ K,
    const float* __restrict__ V, float* __restrict__ O)
{
    __shared__ _Float16 Kt[KTILE * LDK];     // [key][d]  row-major
    __shared__ _Float16 Vt[HDIM * LDV];      // [d][key]  TRANSPOSED
    __shared__ _Float16 Pt[4][16 * LDP];     // per-wave P scratch

    const int qt   = blockIdx.x;
    const int head = blockIdx.y;
    const int tid  = threadIdx.x;
    const int wave = tid >> 5;
    const int lane = tid & 31;
    const int hi   = lane >> 4;
    const int ln   = lane & 15;

    const int qbase = qt * QTILE;
    const int qrow  = qbase + wave * 16;

    const float qk_scale = 0.08838834764831845f * 1.44269504088896341f; // (1/sqrt(128))*log2(e)

    // ---- Q fragments (A layout), fp32 -> f16, scale*log2e folded in ----
    F16Frag qa[4];
    {
        const float* qp = Q + ((size_t)(qrow + ln) * NHEAD + head) * HDIM;
        #pragma unroll
        for (int kk = 0; kk < 4; ++kk)
            #pragma unroll
            for (int h = 0; h < 16; ++h) {
                int kd = kk * 32 + hi * 16 + h;
                qa[kk].e[h] = (_Float16)(qp[kd] * qk_scale);
            }
    }

    F32Frag o[8];
    #pragma unroll
    for (int t = 0; t < 8; ++t) o[t].v = {};
    float mrow[8], lrow[8];
    #pragma unroll
    for (int r = 0; r < 8; ++r) { mrow[r] = -INFINITY; lrow[r] = 0.0f; }

    const int jtiles = (qbase + QTILE) / KTILE;  // causal bound (block level)
    for (int jt = 0; jt < jtiles; ++jt) {
        const int jb = jt * KTILE;

        __syncthreads();   // previous-tile reads complete before overwrite

        // ---- K staging: float4 global loads -> b64 LDS stores (row-major) ----
        #pragma unroll
        for (int i = 0; i < 8; ++i) {
            int idx4 = tid + i * 128;          // quad index over 32x(128/4)
            int kr   = idx4 >> 5;              // key row 0..31
            int kd   = (idx4 & 31) * 4;        // d
            const float4 kv = *(const float4*)(K + ((size_t)(jb + kr) * NHEAD + head) * HDIM + kd);
            _Float16* dst = &Kt[kr * LDK + kd];
            dst[0] = (_Float16)kv.x; dst[1] = (_Float16)kv.y;
            dst[2] = (_Float16)kv.z; dst[3] = (_Float16)kv.w;
        }
        // ---- V staging: scalar, TRANSPOSED store (conflict-free: 18-bank row step) ----
        #pragma unroll
        for (int i = 0; i < 32; ++i) {
            int idx = tid + i * 128;
            int kr  = idx >> 7;                // key row 0..31
            int kd  = idx & 127;               // d
            Vt[kd * LDV + kr] = (_Float16)V[((size_t)(jb + kr) * NHEAD + head) * HDIM + kd];
        }

        // ---- prefetch next KV tile (one 128B cacheline per thread per tensor) ----
        if (jt + 1 < jtiles) {
            int jn = jb + KTILE;
            size_t row = ((size_t)(jn + (tid >> 2)) * NHEAD + head) * HDIM;
            __builtin_prefetch((const char*)(K + row) + (tid & 3) * 128, 0, 0);
            __builtin_prefetch((const char*)(V + row) + (tid & 3) * 128, 0, 0);
        }

        __syncthreads();

        if (jb > qrow + 15) continue;   // tile fully masked for this wave (uniform)

        // ---- S = Q K^T : two 16x16 score tiles ----
        F32Frag s0, s1;
        s0.v = {}; s1.v = {};
        #pragma unroll
        for (int kk = 0; kk < 4; ++kk) {
            F16Frag bk0, bk1;
            const _Float16* kp0 = &Kt[ln        * LDK + kk * 32 + hi * 16];
            const _Float16* kp1 = &Kt[(16 + ln) * LDK + kk * 32 + hi * 16];
            #pragma unroll
            for (int h = 0; h < 16; ++h) { bk0.e[h] = kp0[h]; bk1.e[h] = kp1[h]; }
            s0.v = __builtin_amdgcn_wmma_f32_16x16x32_f16(false, qa[kk].v, false, bk0.v,
                                                          (short)0, s0.v, false, false);
            s1.v = __builtin_amdgcn_wmma_f32_16x16x32_f16(false, qa[kk].v, false, bk1.v,
                                                          (short)0, s1.v, false, false);
        }

        // ---- online softmax: masked only on diagonal tiles (uniform branch) ----
        if (jb + KTILE - 1 > qrow)
            online_update<true >(s0, s1, mrow, lrow, o, Pt[wave], qrow, jb, hi, ln);
        else
            online_update<false>(s0, s1, mrow, lrow, o, Pt[wave], qrow, jb, hi, ln);

        // ---- reload P as 16x32 A fragment (wave-local LDS, in-order DS) ----
        F16Frag pa;
        {
            const _Float16* pp = &Pt[wave][ln * LDP];
            #pragma unroll
            for (int h = 0; h < 8; ++h) pa.e[h]     = pp[hi * 16 + h];
            #pragma unroll
            for (int h = 0; h < 8; ++h) pa.e[8 + h] = pp[hi * 16 + 8 + h];
        }

        // ---- O += P V : transposed-V fragments are contiguous 16-half runs ----
        #pragma unroll
        for (int t = 0; t < 8; ++t) {
            F16Frag bv;
            const _Float16* vp = &Vt[(t * 16 + ln) * LDV + hi * 16];
            #pragma unroll
            for (int h = 0; h < 16; ++h) bv.e[h] = vp[h];
            o[t].v = __builtin_amdgcn_wmma_f32_16x16x32_f16(false, pa.v, false, bv.v,
                                                            (short)0, o[t].v, false, false);
        }
    }

    // ---- epilogue ----
    #pragma unroll
    for (int r = 0; r < 8; ++r) {
        float inv = 1.0f / lrow[r];
        int qg = qrow + hi * 8 + r;
        float* op = O + ((size_t)qg * NHEAD + head) * HDIM;
        #pragma unroll
        for (int t = 0; t < 8; ++t)
            op[t * 16 + ln] = o[t].e[r] * inv;
    }
}

extern "C" void kernel_launch(void* const* d_in, const int* in_sizes, int n_in,
                              void* d_out, int out_size, void* d_ws, size_t ws_size,
                              hipStream_t stream) {
    (void)in_sizes; (void)n_in; (void)out_size; (void)d_ws; (void)ws_size;
    const float* Q = (const float*)d_in[0];
    const float* K = (const float*)d_in[1];
    const float* V = (const float*)d_in[2];
    float* Out = (float*)d_out;
    dim3 grid(S_LEN / QTILE, NHEAD);
    fa_fwd_kernel<<<grid, 128, 0, stream>>>(Q, K, V, Out);
}
